// Performer_31714038513715
// MI455X (gfx1250) — matrix operand order repaired and checked by reading, not compiled, not run on hardware
//
#include <hip/hip_runtime.h>
#include <math.h>
#include <stdint.h>

typedef __attribute__((ext_vector_type(16))) _Float16 v16h;
typedef __attribute__((ext_vector_type(8)))  _Float16 v8h;
typedef __attribute__((ext_vector_type(8)))  float    v8f;

#define B_   4
#define N_   16384
#define DIM_ 384
#define H_   8
#define C_   48
#define M_   24
#define MP_  32      // feature dim padded to WMMA K
#define CE_  64      // ktv cols: 48 v-cols + 1 ones-col (k_sum/D) + pad
#define EPS_ 1e-6f

static __device__ __forceinline__ v8f zero8() {
  v8f z;
#pragma unroll
  for (int i = 0; i < 8; i++) z[i] = 0.0f;
  return z;
}

static __device__ __forceinline__ v8f wmma_f16(v16h a, v16h b, v8f c) {
  // D = A(16x32 f16) * B(32x16 f16) + C(16x16 f32)
  return __builtin_amdgcn_wmma_f32_16x16x32_f16(false, a, false, b,
                                                (short)0, c, false, false);
}

static __device__ __forceinline__ void put4(v16h& f, int i, float4 v) {
  f[i + 0] = (_Float16)v.x; f[i + 1] = (_Float16)v.y;
  f[i + 2] = (_Float16)v.z; f[i + 3] = (_Float16)v.w;
}

// Async 16-byte global -> LDS copy (per-lane), tracked by ASYNCcnt.
static __device__ __forceinline__ void async_copy_b128(const void* gsrc,
                                                       void* ldst) {
  unsigned           lds_off = (unsigned)(uintptr_t)ldst;       // addr[31:0] = LDS offset
  unsigned long long gaddr   = (unsigned long long)(uintptr_t)gsrc;
  asm volatile("global_load_async_to_lds_b128 %0, %1, off"
               :: "v"(lds_off), "v"(gaddr) : "memory");
}

static __device__ __forceinline__ void wait_async0() {
  asm volatile("s_wait_asynccnt 0x0" ::: "memory");
}

// ---------------------------------------------------------------------------
// Kernel 0: zero the ktv accumulator
// ---------------------------------------------------------------------------
__global__ void perf_zero(float* __restrict__ p, int n) {
  int i = blockIdx.x * blockDim.x + threadIdx.x;
  if (i < n) p[i] = 0.0f;
}

// ---------------------------------------------------------------------------
// Kernel 1: per (b,h,128-row tile): QKV GEMM -> feature map -> ktv reduction
// ---------------------------------------------------------------------------
__global__ __launch_bounds__(256)
void perf_qkv_phi_ktv(const float* __restrict__ x,
                      const float* __restrict__ qkv_w,
                      const float* __restrict__ w,
                      _Float16* __restrict__ phiQ,   // [B*H, N, MP] f16
                      float* __restrict__ ktv)       // [B*H, MP, CE] f32
{
  // Pooled LDS (55040 B). wbuf (18 KB, live only in the K loop) aliases
  // qs+ks (28 KB, live only after it); the loop's trailing barrier
  // separates the lifetimes.
  __shared__ __align__(16) char smem[55040];
  float    (*wbuf)[32]  = (float(*)[32])     &smem[0];      // [144][32] f32
  _Float16 (*qs)[56]    = (_Float16(*)[56])  &smem[0];      // [128][56]
  _Float16 (*ks)[56]    = (_Float16(*)[56])  &smem[14336];  // [128][56]
  _Float16 (*vT)[132]   = (_Float16(*)[132]) &smem[28672];  // [64][132]
  _Float16 (*pkT)[132]  = (_Float16(*)[132]) &smem[45568];  // [32][132]
  float*    sq          = (float*)&smem[54016];             // [128]
  float*    sk          = sq + 128;                          // [128]

  const int bh  = blockIdx.y;
  const int b   = bh / H_;
  const int h   = bh % H_;
  const int n0  = blockIdx.x * 128;

  const int tid  = threadIdx.x;
  const int wave = tid >> 5;
  const int lane = tid & 31;
  const int half = lane >> 4;
  const int lr   = lane & 15;

  const float kscale   = 0.3799178428257963f;  // 48^-0.25
  const float invSqrtM = 0.2041241452319315f;  // 1/sqrt(24)

  // ---- Stage 1: QKV GEMM, 144 cols = [q(48)|k(48)|v(48)], K = 384 ----
  v8f acc[9];
#pragma unroll
  for (int t = 0; t < 9; t++) acc[t] = zero8();

  const int    rowA = n0 + wave * 16 + lr;
  const float* xrow = x + ((size_t)b * N_ + rowA) * DIM_;

  for (int kk = 0; kk < DIM_ / 32; ++kk) {
    const int k0 = kk * 32;

    // Cooperative async stage of W_h[:, k0:k0+32] (144x32 f32) into LDS:
    // 1152 float4s across 256 lanes.
#pragma unroll
    for (int it = 0; it < 5; ++it) {
      const int idx = it * 256 + tid;
      if (idx < 1152) {
        const int row     = idx >> 3;   // 0..143
        const int quad    = idx & 7;    // float4 within the 32-wide slab
        const int section = row / 48;
        const int jj      = row - section * 48;
        const float* src  = qkv_w +
            (size_t)(section * DIM_ + h * C_ + jj) * DIM_ + k0 + quad * 4;
        async_copy_b128(src, &wbuf[row][quad * 4]);
      }
    }
    wait_async0();
    __syncthreads();

    __builtin_prefetch(xrow + k0 + 32, 0, 3);
    v16h afrag;
    put4(afrag, 0,  *(const float4*)(xrow + k0 + 16 * half));
    put4(afrag, 4,  *(const float4*)(xrow + k0 + 16 * half + 4));
    put4(afrag, 8,  *(const float4*)(xrow + k0 + 16 * half + 8));
    put4(afrag, 12, *(const float4*)(xrow + k0 + 16 * half + 12));

#pragma unroll
    for (int t = 0; t < 9; t++) {
      const float* wr = &wbuf[t * 16 + lr][16 * half];
      v16h bfrag;
      put4(bfrag, 0,  *(const float4*)(wr + 0));
      put4(bfrag, 4,  *(const float4*)(wr + 4));
      put4(bfrag, 8,  *(const float4*)(wr + 8));
      put4(bfrag, 12, *(const float4*)(wr + 12));
      acc[t] = wmma_f16(afrag, bfrag, acc[t]);
    }
    __syncthreads();   // wbuf consumed; safe to overwrite next iter / alias qs
  }

  // Stash q*scale, k*scale, v^T into LDS (C-frag: col=lr, row=v+8*half)
#pragma unroll
  for (int t = 0; t < 3; t++) {
#pragma unroll
    for (int v = 0; v < 8; v++) {
      const int rl = wave * 16 + v + 8 * half;
      qs[rl][t * 16 + lr] = (_Float16)(acc[t][v]     * kscale);
      ks[rl][t * 16 + lr] = (_Float16)(acc[3 + t][v] * kscale);
      vT[t * 16 + lr][rl] = (_Float16)(acc[6 + t][v]);
    }
  }
  if (tid < 128) {
    vT[48][tid] = (_Float16)1.0f;   // ones column -> k_sum / D
#pragma unroll
    for (int c = 49; c < CE_; c++) vT[c][tid] = (_Float16)0.0f;
  }
  __syncthreads();

  // ---- Stage 2: row norms ||t*scale||^2 ----
  {
    const int n = tid & 127;
    float s = 0.0f;
    if (tid < 128) {
      for (int c = 0; c < C_; c++) { float q = (float)qs[n][c]; s += q * q; }
      sq[n] = s;
    } else {
      for (int c = 0; c < C_; c++) { float q = (float)ks[n][c]; s += q * q; }
      sk[n] = s;
    }
  }
  __syncthreads();

  // ---- Stage 3: feature-map GEMM logits = (t*scale) @ w[h]^T, K=48 pad 64 ----
  v16h bw[2][2];
#pragma unroll
  for (int mt = 0; mt < 2; mt++) {
    const int m = mt * 16 + lr;
#pragma unroll
    for (int kstep = 0; kstep < 2; kstep++) {
      v16h f;
#pragma unroll
      for (int i = 0; i < 16; i++) {
        const int c = kstep * 32 + 16 * half + i;
        const float val =
            (m < M_ && c < C_) ? w[((size_t)h * M_ + m) * C_ + c] : 0.0f;
        f[i] = (_Float16)val;
      }
      bw[mt][kstep] = f;
    }
  }

  v8f accq[2], acck[2];
  accq[0] = zero8(); accq[1] = zero8();
  acck[0] = zero8(); acck[1] = zero8();
#pragma unroll
  for (int kstep = 0; kstep < 2; kstep++) {
    const int k0 = kstep * 32;
    const int rl = wave * 16 + lr;
    v16h aq, ak;
#pragma unroll
    for (int i = 0; i < 16; i++) {
      const int c = k0 + 16 * half + i;
      aq[i] = (c < C_) ? qs[rl][c] : (_Float16)0.0f;
      ak[i] = (c < C_) ? ks[rl][c] : (_Float16)0.0f;
    }
#pragma unroll
    for (int mt = 0; mt < 2; mt++) {
      accq[mt] = wmma_f16(aq, bw[mt][kstep], accq[mt]);
      acck[mt] = wmma_f16(ak, bw[mt][kstep], acck[mt]);
    }
  }

  // exp transform; phi_q -> global, phi_k^T -> LDS
#pragma unroll
  for (int mt = 0; mt < 2; mt++) {
    const int m = mt * 16 + lr;
#pragma unroll
    for (int v = 0; v < 8; v++) {
      const int rl = wave * 16 + v + 8 * half;
      const int n  = n0 + rl;
      float pq = 0.0f, pk = 0.0f;
      if (m < M_) {
        pq = __expf(accq[mt][v] - 0.5f * sq[rl]) * invSqrtM;
        pk = __expf(acck[mt][v] - 0.5f * sk[rl]) * invSqrtM;
      }
      phiQ[((size_t)bh * N_ + n) * MP_ + m] = (_Float16)pq;
      pkT[m][rl] = (_Float16)pk;
    }
  }
  __syncthreads();

  // ---- Stage 4: ktv += phi_k^T @ [v | 1 | 0], K = 128 rows ----
  {
    const int mt = wave >> 2;   // 0..1 : m tile
    const int ct = wave & 3;    // 0..3 : c tile
    v8f a9 = zero8();
#pragma unroll
    for (int kstep = 0; kstep < 4; kstep++) {
      const int k0 = kstep * 32;
      const int m  = mt * 16 + lr;
      const int c  = ct * 16 + lr;
      v16h afr, bfr;
#pragma unroll
      for (int i = 0; i < 16; i++) {
        const int nn = k0 + 16 * half + i;
        afr[i] = pkT[m][nn];
      }
#pragma unroll
      for (int i = 0; i < 16; i++) bfr[i] = vT[c][k0 + 16 * half + i];
      a9 = wmma_f16(afr, bfr, a9);
    }
    const int c = ct * 16 + lr;
#pragma unroll
    for (int v = 0; v < 8; v++) {
      const int m = mt * 16 + v + 8 * half;
      atomicAdd(&ktv[((size_t)bh * MP_ + m) * CE_ + c], a9[v]);
    }
  }
}

// ---------------------------------------------------------------------------
// Kernel 2: out_head = (phi_q @ ktv) / D ; D comes out of column 48
// ---------------------------------------------------------------------------
__global__ __launch_bounds__(256)
void perf_attn_out(const _Float16* __restrict__ phiQ,
                   const float* __restrict__ ktv,
                   _Float16* __restrict__ headOut)   // [B, N, DIM] f16
{
  __shared__ float dsh[128];
  const int bh = blockIdx.y;
  const int b  = bh / H_;
  const int h  = bh % H_;
  const int n0 = blockIdx.x * 128;
  const int tid = threadIdx.x, wave = tid >> 5, lane = tid & 31;
  const int half = lane >> 4, lr = lane & 15;

  v16h afr;
  {
    const _Float16* pr = phiQ + ((size_t)bh * N_ + n0 + wave * 16 + lr) * MP_;
    const v8h lo = *(const v8h*)(pr + 16 * half);
    const v8h hi = *(const v8h*)(pr + 16 * half + 8);
#pragma unroll
    for (int i = 0; i < 8; i++) { afr[i] = lo[i]; afr[8 + i] = hi[i]; }
  }

  v8f acc[4];
#pragma unroll
  for (int ct = 0; ct < 4; ct++) {
    const int c = ct * 16 + lr;
    v16h bfr;
#pragma unroll
    for (int i = 0; i < 16; i++) {
      const int m = 16 * half + i;
      bfr[i] = (_Float16)ktv[((size_t)bh * MP_ + m) * CE_ + c];
    }
    acc[ct] = wmma_f16(afr, bfr, zero8());
  }

  if (lr == 0) {   // column 48 = D
#pragma unroll
    for (int v = 0; v < 8; v++)
      dsh[wave * 16 + v + 8 * half] = acc[3][v] + EPS_;
  }
  __syncthreads();

#pragma unroll
  for (int ct = 0; ct < 3; ct++) {
    const int cg = h * C_ + ct * 16 + lr;
#pragma unroll
    for (int v = 0; v < 8; v++) {
      const int rl = wave * 16 + v + 8 * half;
      const int n  = n0 + rl;
      const float val = acc[ct][v] / dsh[rl];
      headOut[((size_t)b * N_ + n) * DIM_ + cg] = (_Float16)val;
    }
  }
}

// ---------------------------------------------------------------------------
// Kernel 3: final projection  out = headOut @ proj_w^T + proj_b  (f32 out)
// ---------------------------------------------------------------------------
__global__ __launch_bounds__(256)
void perf_proj(const _Float16* __restrict__ headOut,
               const float* __restrict__ proj_w,
               const float* __restrict__ proj_b,
               float* __restrict__ out)
{
  __shared__ __align__(16) float wbufC[64][32];   // proj_w K-slab, 8 KB

  const int r0 = blockIdx.x * 128;
  const int c0 = blockIdx.y * 64;
  const int tid = threadIdx.x, wave = tid >> 5, lane = tid & 31;
  const int half = lane >> 4, lr = lane & 15;
  const int r = r0 + wave * 16 + lr;
  const _Float16* arow = headOut + (size_t)r * DIM_;

  v8f acc[4];
#pragma unroll
  for (int t = 0; t < 4; t++) acc[t] = zero8();

  for (int kk = 0; kk < DIM_ / 32; ++kk) {
    const int k0 = kk * 32;

    // Cooperative async stage of proj_w[c0:c0+64, k0:k0+32]: 512 float4s.
#pragma unroll
    for (int it = 0; it < 2; ++it) {
      const int idx  = it * 256 + tid;   // 0..511
      const int row  = idx >> 3;         // 0..63
      const int quad = idx & 7;
      const float* src = proj_w + (size_t)(c0 + row) * DIM_ + k0 + quad * 4;
      async_copy_b128(src, &wbufC[row][quad * 4]);
    }
    wait_async0();
    __syncthreads();

    __builtin_prefetch(arow + k0 + 32, 0, 3);
    v16h afr;
    {
      const v8h lo = *(const v8h*)(arow + k0 + 16 * half);
      const v8h hi = *(const v8h*)(arow + k0 + 16 * half + 8);
#pragma unroll
      for (int i = 0; i < 8; i++) { afr[i] = lo[i]; afr[8 + i] = hi[i]; }
    }
#pragma unroll
    for (int ct = 0; ct < 4; ct++) {
      const float* wr = &wbufC[ct * 16 + lr][16 * half];
      v16h bfr;
      put4(bfr, 0,  *(const float4*)(wr + 0));
      put4(bfr, 4,  *(const float4*)(wr + 4));
      put4(bfr, 8,  *(const float4*)(wr + 8));
      put4(bfr, 12, *(const float4*)(wr + 12));
      acc[ct] = wmma_f16(afr, bfr, acc[ct]);
    }
    __syncthreads();
  }

#pragma unroll
  for (int ct = 0; ct < 4; ct++) {
    const int j = c0 + ct * 16 + lr;
    const float bias = proj_b[j];
#pragma unroll
    for (int v = 0; v < 8; v++) {
      const int rr = r0 + wave * 16 + v + 8 * half;
      out[(size_t)rr * DIM_ + j] = acc[ct][v] + bias;
    }
  }
}

// ---------------------------------------------------------------------------
extern "C" void kernel_launch(void* const* d_in, const int* in_sizes, int n_in,
                              void* d_out, int out_size, void* d_ws, size_t ws_size,
                              hipStream_t stream)
{
  (void)in_sizes; (void)n_in; (void)out_size; (void)ws_size;
  const float* x      = (const float*)d_in[0];
  const float* qkv_w  = (const float*)d_in[1];
  const float* proj_w = (const float*)d_in[2];
  const float* proj_b = (const float*)d_in[3];
  const float* w      = (const float*)d_in[4];
  float* out = (float*)d_out;

  char* ws = (char*)d_ws;
  float*    ktv     = (float*)ws;                      // 32*32*64*4 = 256 KB
  _Float16* phiQ    = (_Float16*)(ws + (size_t)(1 << 18));
  _Float16* headOut = (_Float16*)(ws + (size_t)(1 << 18) +
                                  (size_t)B_ * H_ * N_ * MP_ * sizeof(_Float16));

  const int ktvN = B_ * H_ * MP_ * CE_;
  perf_zero<<<(ktvN + 255) / 256, 256, 0, stream>>>(ktv, ktvN);

  dim3 gA(N_ / 128, B_ * H_);
  perf_qkv_phi_ktv<<<gA, 256, 0, stream>>>(x, qkv_w, w, phiQ, ktv);
  perf_attn_out<<<gA, 256, 0, stream>>>(phiQ, ktv, headOut);

  dim3 gC((B_ * N_) / 128, DIM_ / 64);
  perf_proj<<<gC, 256, 0, stream>>>(headOut, proj_w, proj_b, out);
}
